// GNNAutoEncoder_65712999629271
// MI455X (gfx1250) — matrix operand order repaired and checked by reading, not compiled, not run on hardware
//
#include <hip/hip_runtime.h>
#include <hip/hip_bf16.h>
#include <math.h>

typedef __attribute__((ext_vector_type(16))) _Float16 v16h;
typedef __attribute__((ext_vector_type(8)))  _Float16 v8h;
typedef __attribute__((ext_vector_type(8)))  float    v8f;
typedef __attribute__((ext_vector_type(4)))  unsigned int u32x4v;
typedef __attribute__((ext_vector_type(8)))  int i32x8;
typedef __attribute__((ext_vector_type(4)))  int i32x4;

union frag16 { v16h v; v8h h[2]; };

#define TILE   128
#define KCHUNK 64
#define LDAH   72   // 64 halves + 4-dword TDM pad per row -> conflict-free LDS

// -------- Tensor Data Mover: 2D tile (128 rows x 64 halves, row stride K) ---
// D# per CDNA5 ISA ch.8: group0 {count, lds_addr, global_addr, type=2},
// group1 {data_size=2B, pad 32dw->+4dw, tensor/tile dims, dim0 stride}.
__device__ __forceinline__ unsigned int lds_off(const void* p)
{
    // LDS aperture keeps the byte offset in addr[31:0]
    return (unsigned int)(unsigned long long)(uintptr_t)p;
}

__device__ __forceinline__ void tdm_load_tile(const _Float16* gsrc,
                                              unsigned int lds_byte,
                                              int row_stride_elems)
{
    unsigned long long ga = (unsigned long long)(uintptr_t)gsrc;
    u32x4v g0;
    g0[0] = 1u;                                       // count=1, user descriptor
    g0[1] = lds_byte;                                 // lds_addr
    g0[2] = (unsigned int)(ga & 0xFFFFFFFFu);         // global_addr[31:0]
    g0[3] = (unsigned int)((ga >> 32) & 0x01FFFFFFu)  // global_addr[56:32]
          | (2u << 30);                               // type = 2 ("image")
    i32x8 g1;
    g1[0] = (int)((1u << 16)      // data_size = 2 bytes
                | (1u << 20)      // pad_enable
                | (4u << 22)      // pad_interval: 32 dwords (= one 64-half row)
                | (3u << 25));    // pad_amount:   4 dwords  (= 8 halves)
    g1[1] = (int)(64u  << 16);    // tensor_dim0[15:0] = 64   (bits 63:48)
    g1[2] = (int)(128u << 16);    // tensor_dim1[15:0] = 128  (bits 95:80)
    g1[3] = (int)(64u  << 16);    // tile_dim0 = 64           (bits 127:112)
    g1[4] = 128;                  // tile_dim1 = 128          (bits 143:128)
    g1[5] = row_stride_elems;     // tensor_dim0_stride[31:0] (bits 191:160)
    g1[6] = 0;
    g1[7] = 0;
    i32x4 gz = {0, 0, 0, 0};      // groups 2/3 unused (2D tensor)
    asm volatile("tensor_load_to_lds %0, %1, %2, %3"
                 :
                 : "s"(g0), "s"(g1), "s"(gz), "s"(gz)
                 : "memory");
}

// -------------------- WMMA GEMM-NT: C[M,N] = A[M,K] * B[N,K]^T --------------
// TDM double-buffered staging; mode 0: store f32, mode 1: sigmoid then store.
__global__ void __launch_bounds__(256)
gemm_nt_wmma(const _Float16* __restrict__ A, const _Float16* __restrict__ B,
             float* __restrict__ C, int M, int N, int K, int mode)
{
    __shared__ _Float16 sh[2][2][TILE][LDAH];   // [buf][A|B][row][col] = 72 KB

    const int tid  = threadIdx.x;
    const int w    = tid >> 5;          // wave 0..7
    const int lane = tid & 31;
    const int lr   = lane & 15;
    const int hi   = lane >> 4;         // 0 / 1
    const int i0   = blockIdx.x * TILE;
    const int j0   = blockIdx.y * TILE;
    const int nchunks = K / KCHUNK;

    v8f zero = {};
    v8f acc[8];
#pragma unroll
    for (int t = 0; t < 8; ++t) acc[t] = zero;

    if (w == 0) {   // TDM ops ignore EXEC; issued once per wave
        tdm_load_tile(A + (size_t)i0 * K, lds_off(&sh[0][0][0][0]), K);
        tdm_load_tile(B + (size_t)j0 * K, lds_off(&sh[0][1][0][0]), K);
    }

    for (int c = 0; c < nchunks; ++c) {
        const int cur = c & 1;
        if (w == 0) {
            if (c + 1 < nchunks) {
                const int kc2 = (c + 1) * KCHUNK;
                const int nxt = cur ^ 1;
                tdm_load_tile(A + (size_t)i0 * K + kc2,
                              lds_off(&sh[nxt][0][0][0]), K);
                tdm_load_tile(B + (size_t)j0 * K + kc2,
                              lds_off(&sh[nxt][1][0][0]), K);
                __builtin_amdgcn_s_wait_tensorcnt(2);  // current pair done
            } else {
                __builtin_amdgcn_s_wait_tensorcnt(0);
            }
        }
        __syncthreads();

#pragma unroll
        for (int ks = 0; ks < 2; ++ks) {
            const int k0 = ks * 32;
            // A fragment: 16x32 f16 (ISA 7.12.2: lo lanes K{0..7,16..23},
            // hi lanes K{8..15,24..31})
            frag16 af;
            {
                const _Float16* pa = &sh[cur][0][w * 16 + lr][k0 + hi * 8];
                af.h[0] = *(const v8h*)(pa);
                af.h[1] = *(const v8h*)(pa + 16);
            }
            // preload all 8 B fragments, then back-to-back WMMA chain
            frag16 bf[8];
#pragma unroll
            for (int tj = 0; tj < 8; ++tj) {
                const _Float16* pb = &sh[cur][1][tj * 16 + lr][k0 + hi * 8];
                bf[tj].h[0] = *(const v8h*)(pb);
                bf[tj].h[1] = *(const v8h*)(pb + 16);
            }
#pragma unroll
            for (int tj = 0; tj < 8; ++tj) {
                acc[tj] = __builtin_amdgcn_wmma_f32_16x16x32_f16(
                    false, af.v, false, bf[tj].v, (short)0, acc[tj],
                    false, false);
            }
        }
        __syncthreads();
    }

#pragma unroll
    for (int tj = 0; tj < 8; ++tj) {
        const int ocol = j0 + tj * 16 + lr;
#pragma unroll
        for (int r = 0; r < 8; ++r) {
            const int orow = i0 + w * 16 + r + hi * 8;  // C layout: VGPR r -> M=r / r+8
            float v = acc[tj][r];
            if (mode == 1) v = 1.0f / (1.0f + __expf(-v));
            C[(size_t)orow * N + ocol] = v;
        }
    }
}

// -------------------- helper kernels ---------------------------------------
__global__ void f32_to_f16_k(const float* __restrict__ in, _Float16* __restrict__ out, int n)
{
    int i = blockIdx.x * blockDim.x + threadIdx.x;
    if (i < n) out[i] = (_Float16)in[i];
}

__global__ void transpose_w_k(const float* __restrict__ W, _Float16* __restrict__ Wt,
                              int K, int Nw)   // W[K,Nw] -> Wt[Nw,K]
{
    int i = blockIdx.x * blockDim.x + threadIdx.x;
    if (i >= K * Nw) return;
    int k = i / Nw, j = i % Nw;
    Wt[(size_t)j * K + k] = (_Float16)W[(size_t)k * Nw + j];
}

__global__ void fill_f32_k(float* __restrict__ p, float v, int n)
{
    int i = blockIdx.x * blockDim.x + threadIdx.x;
    if (i < n) p[i] = v;
}

__global__ void deg_count_k(const long long* __restrict__ ei, int E, float* __restrict__ deg)
{
    int e = blockIdx.x * blockDim.x + threadIdx.x;
    if (e < E) atomicAdd(&deg[(int)ei[(size_t)E + e]], 1.0f);
}

__global__ void dinv_k(const float* __restrict__ deg, float* __restrict__ dinv, int n)
{
    int i = blockIdx.x * blockDim.x + threadIdx.x;
    if (i < n) dinv[i] = rsqrtf(deg[i]);
}

__global__ void edge_prep_k(const long long* __restrict__ ei, int E,
                            const float* __restrict__ dinv,
                            int* __restrict__ s32, int* __restrict__ d32,
                            float* __restrict__ norm)
{
    int e = blockIdx.x * blockDim.x + threadIdx.x;
    if (e >= E) return;
    int s = (int)ei[e];
    int d = (int)ei[(size_t)E + e];
    s32[e] = s;
    d32[e] = d;
    norm[e] = dinv[s] * dinv[d];
}

__global__ void agg_k(const int* __restrict__ s32, const int* __restrict__ d32,
                      const float* __restrict__ norm, const float* __restrict__ h,
                      float* __restrict__ agg, int E)
{
    int tid = blockIdx.x * blockDim.x + threadIdx.x;
    if (tid >= E * 128) return;
    int e = tid >> 7;
    int f = tid & 127;
    float v = h[(size_t)s32[e] * 128 + f] * norm[e];
    atomicAdd(&agg[(size_t)d32[e] * 128 + f], v);
}

__global__ void z_k(const float* __restrict__ agg, const float* __restrict__ h,
                    const float* __restrict__ dinv, const float* __restrict__ bias,
                    float* __restrict__ z32, _Float16* __restrict__ z16, int n)
{
    int tid = blockIdx.x * blockDim.x + threadIdx.x;
    if (tid >= n) return;
    int node = tid >> 7;
    int f = tid & 127;
    float di = dinv[node];
    float v = agg[tid] + di * di * h[tid] + bias[f];   // self-loop fused here
    v = fmaxf(v, 0.0f);
    z32[tid] = v;
    z16[tid] = (_Float16)v;
}

// -------------------- launch -----------------------------------------------
extern "C" void kernel_launch(void* const* d_in, const int* in_sizes, int n_in,
                              void* d_out, int out_size, void* d_ws, size_t ws_size,
                              hipStream_t stream)
{
    (void)n_in; (void)out_size; (void)ws_size;

    const int NN   = 8192;   // nodes
    const int FIN  = 256;    // input features
    const int FOUT = 128;    // hidden features
    const int E    = in_sizes[1] / 2;

    const float*     X    = (const float*)d_in[0];
    const long long* EI   = (const long long*)d_in[1];   // int64 edge_index [2,E]
    const float*     W    = (const float*)d_in[2];
    const float*     bias = (const float*)d_in[3];

    // workspace layout (all chunks naturally aligned)
    char* p = (char*)d_ws;
    _Float16* Xh   = (_Float16*)p;  p += (size_t)NN * FIN * sizeof(_Float16);
    _Float16* Wt   = (_Float16*)p;  p += (size_t)FOUT * FIN * sizeof(_Float16);
    float*    h    = (float*)p;     p += (size_t)NN * FOUT * sizeof(float);
    float*    agg  = (float*)p;     p += (size_t)NN * FOUT * sizeof(float);
    float*    deg  = (float*)p;     p += (size_t)NN * sizeof(float);
    float*    dinv = (float*)p;     p += (size_t)NN * sizeof(float);
    int*      s32  = (int*)p;       p += (size_t)E * sizeof(int);
    int*      d32  = (int*)p;       p += (size_t)E * sizeof(int);
    float*    norm = (float*)p;     p += (size_t)E * sizeof(float);
    _Float16* zh   = (_Float16*)p;  p += (size_t)NN * FOUT * sizeof(_Float16);

    float* adj = (float*)d_out;                       // [NN, NN]
    float* z32 = adj + (size_t)NN * NN;               // [NN, FOUT]

    const int T = 256;

    // 1) precision conversion for WMMA operands
    f32_to_f16_k<<<(NN * FIN + T - 1) / T, T, 0, stream>>>(X, Xh, NN * FIN);
    transpose_w_k<<<(FIN * FOUT + T - 1) / T, T, 0, stream>>>(W, Wt, FIN, FOUT);

    // 2) h = X @ W  (WMMA, f32 accum, TDM-staged)
    dim3 g1(NN / TILE, FOUT / TILE);
    gemm_nt_wmma<<<g1, T, 0, stream>>>(Xh, Wt, h, NN, FOUT, FIN, 0);

    // 3) degrees + symmetric norm
    fill_f32_k<<<(NN + T - 1) / T, T, 0, stream>>>(deg, 1.0f, NN);   // self loops
    deg_count_k<<<(E + T - 1) / T, T, 0, stream>>>(EI, E, deg);
    dinv_k<<<(NN + T - 1) / T, T, 0, stream>>>(deg, dinv, NN);
    edge_prep_k<<<(E + T - 1) / T, T, 0, stream>>>(EI, E, dinv, s32, d32, norm);

    // 4) message aggregation (scatter-add)
    fill_f32_k<<<(NN * FOUT + T - 1) / T, T, 0, stream>>>(agg, 0.0f, NN * FOUT);
    agg_k<<<(E * FOUT + T - 1) / T, T, 0, stream>>>(s32, d32, norm, h, agg, E);

    // 5) z = relu(agg + dinv^2*h + b); emit f32 (output) and f16 (for GEMM2)
    z_k<<<(NN * FOUT + T - 1) / T, T, 0, stream>>>(agg, h, dinv, bias, z32, zh, NN * FOUT);

    // 6) adj = sigmoid(z @ z^T)  (dominant 17.2 GFLOP GEMM, WMMA + fused sigmoid)
    dim3 g2(NN / TILE, NN / TILE);
    gemm_nt_wmma<<<g2, T, 0, stream>>>(zh, zh, adj, NN, NN, FOUT, 1);
}
